// ContrastiveLoss_62637803045457
// MI455X (gfx1250) — matrix operand order repaired and checked by reading, not compiled, not run on hardware
//
#include <hip/hip_runtime.h>
#include <hip/hip_bf16.h>

typedef __attribute__((ext_vector_type(16))) _Float16 v16h;
typedef __attribute__((ext_vector_type(8)))  _Float16 v8h;
typedef __attribute__((ext_vector_type(4)))  _Float16 v4h;
typedef __attribute__((ext_vector_type(8)))  float    v8f;

#define BN    8192
#define DK    512
#define INV_T (1.0f / 0.07f)

// ---------------- kernel 1: row-normalize + split fp32 -> f16 hi/lo ----------
__global__ __launch_bounds__(128)
void k_norm_split(const float* __restrict__ X,
                  _Float16* __restrict__ Xhi,
                  _Float16* __restrict__ Xlo) {
    const int row = blockIdx.x;
    const int t   = threadIdx.x;                 // 0..127, 4 floats each
    const float4 v = ((const float4*)(X + (size_t)row * DK))[t];
    float ss = v.x * v.x + v.y * v.y + v.z * v.z + v.w * v.w;
    #pragma unroll
    for (int m = 16; m >= 1; m >>= 1) ss += __shfl_xor(ss, m, 32);
    __shared__ float red[4];
    if ((t & 31) == 0) red[t >> 5] = ss;
    __syncthreads();
    const float tot = red[0] + red[1] + red[2] + red[3];
    const float inv = 1.0f / fmaxf(sqrtf(tot), 1e-12f);

    const float xs0 = v.x * inv, xs1 = v.y * inv, xs2 = v.z * inv, xs3 = v.w * inv;
    v4h hi, lo;
    _Float16 h;
    h = (_Float16)xs0; hi[0] = h; lo[0] = (_Float16)(xs0 - (float)h);
    h = (_Float16)xs1; hi[1] = h; lo[1] = (_Float16)(xs1 - (float)h);
    h = (_Float16)xs2; hi[2] = h; lo[2] = (_Float16)(xs2 - (float)h);
    h = (_Float16)xs3; hi[3] = h; lo[3] = (_Float16)(xs3 - (float)h);
    ((v4h*)(Xhi + (size_t)row * DK))[t] = hi;
    ((v4h*)(Xlo + (size_t)row * DK))[t] = lo;
}

// ---------------- kernel 2: WMMA sim block + exp-sum / pos capture ----------
// WG = 256 threads = 8 waves (wave32). Each WG owns 32 rows (A tile in LDS,
// 64 KB hi+lo). Each wave walks column tiles with stride 128 and computes TWO
// row tiles per B fragment (register blocking): 6 WMMAs per B load pair ->
// 24 FLOP per L2 byte of B traffic (vs 12 before).
__global__ __launch_bounds__(256)
void k_sim_lse(const _Float16* __restrict__ Xhi,
               const _Float16* __restrict__ Xlo,
               float* __restrict__ partials,      // [8][8192]
               float* __restrict__ rowpos) {      // [8192]
    __shared__ _Float16 sAhi[32 * DK];            // 32 KB
    __shared__ _Float16 sAlo[32 * DK];            // 32 KB

    const int tid  = threadIdx.x;
    const int wv   = tid >> 5;        // wave 0..7
    const int lane = tid & 31;
    const int half = lane >> 4;       // 0/1
    const int nl   = lane & 15;       // M for A-frag, N for B/C-frag
    const int rb   = blockIdx.x * 32; // row base

    // Cooperative copy of the contiguous 32-row A block (hi and lo).
    {
        const uint4* sh = (const uint4*)(Xhi + (size_t)rb * DK);
        const uint4* sl = (const uint4*)(Xlo + (size_t)rb * DK);
        uint4* dh = (uint4*)sAhi;
        uint4* dl = (uint4*)sAlo;
        #pragma unroll
        for (int i = 0; i < 8; ++i) {             // 2048 uint4 / 256 threads
            dh[tid + i * 256] = sh[tid + i * 256];
            dl[tid + i * 256] = sl[tid + i * 256];
        }
    }
    __syncthreads();

    float part0[8], part1[8];
    #pragma unroll
    for (int r = 0; r < 8; ++r) { part0[r] = 0.0f; part1[r] = 0.0f; }

    // A-frag (16-bit 16x32): lane M = nl; VGPR v holds K = 2v + 16*half.
    const int abase0 = nl * DK + 16 * half;           // row tile 0: rows rb..rb+15
    const int abase1 = (nl + 16) * DK + 16 * half;    // row tile 1: rows rb+16..rb+31

    for (int cb = wv * 16; cb < BN; cb += 128) {
        const int col = cb + nl;
        const _Float16* bh = Xhi + ((size_t)col << 9) + (half << 4);
        const _Float16* bl = Xlo + ((size_t)col << 9) + (half << 4);
        // Prefetch next column tile's B stream (+128 cols = +128 KB).
        if (cb + 128 < BN) {
            __builtin_prefetch(bh + 128 * DK, 0, 1);
            __builtin_prefetch(bl + 128 * DK, 0, 1);
        }
        v8f acc0 = {};
        v8f acc1 = {};
        #pragma unroll 4
        for (int kc = 0; kc < DK; kc += 32) {
            // B-frag (32x16): lane N = nl; VGPR v holds K = 2v + 16*half
            // -> K = kc+16h .. kc+16h+15 contiguous (32 bytes).
            v16h Bhi = *(const v16h*)(bh + kc);
            v16h Blo = *(const v16h*)(bl + kc);

            const int a0 = abase0 + kc;
            v8h ah0 = *(const v8h*)(sAhi + a0);        // K = kc+16h .. +7
            v8h ah1 = *(const v8h*)(sAhi + a0 + 8);    // K = kc+16h+8 .. +15
            v8h al0 = *(const v8h*)(sAlo + a0);
            v8h al1 = *(const v8h*)(sAlo + a0 + 8);
            v16h Ahi0 = __builtin_shufflevector(ah0, ah1,
                0,1,2,3,4,5,6,7,8,9,10,11,12,13,14,15);
            v16h Alo0 = __builtin_shufflevector(al0, al1,
                0,1,2,3,4,5,6,7,8,9,10,11,12,13,14,15);

            const int a1 = abase1 + kc;
            v8h bh0 = *(const v8h*)(sAhi + a1);
            v8h bh1 = *(const v8h*)(sAhi + a1 + 8);
            v8h bl0 = *(const v8h*)(sAlo + a1);
            v8h bl1 = *(const v8h*)(sAlo + a1 + 8);
            v16h Ahi1 = __builtin_shufflevector(bh0, bh1,
                0,1,2,3,4,5,6,7,8,9,10,11,12,13,14,15);
            v16h Alo1 = __builtin_shufflevector(bl0, bl1,
                0,1,2,3,4,5,6,7,8,9,10,11,12,13,14,15);

            // split-f16 fp32 emulation: hi*hi + hi*lo + lo*hi, two row tiles
            acc0 = __builtin_amdgcn_wmma_f32_16x16x32_f16(
                       false, Ahi0, false, Bhi, (short)0, acc0, false, false);
            acc1 = __builtin_amdgcn_wmma_f32_16x16x32_f16(
                       false, Ahi1, false, Bhi, (short)0, acc1, false, false);
            acc0 = __builtin_amdgcn_wmma_f32_16x16x32_f16(
                       false, Ahi0, false, Blo, (short)0, acc0, false, false);
            acc1 = __builtin_amdgcn_wmma_f32_16x16x32_f16(
                       false, Ahi1, false, Blo, (short)0, acc1, false, false);
            acc0 = __builtin_amdgcn_wmma_f32_16x16x32_f16(
                       false, Alo0, false, Bhi, (short)0, acc0, false, false);
            acc1 = __builtin_amdgcn_wmma_f32_16x16x32_f16(
                       false, Alo1, false, Bhi, (short)0, acc1, false, false);
        }
        // C/D layout: VGPR r -> row M = r + 8*half, col N = nl.
        #pragma unroll
        for (int r = 0; r < 8; ++r) {
            const int i0 = rb + r + 8 * half;
            const float s0 = acc0[r] * INV_T;
            if (col == ((i0 + (BN / 2)) & (BN - 1))) rowpos[i0] = s0;
            part0[r] += (i0 == col) ? 0.0f : __expf(s0);  // diag masked (-inf)

            const int i1 = i0 + 16;
            const float s1 = acc1[r] * INV_T;
            if (col == ((i1 + (BN / 2)) & (BN - 1))) rowpos[i1] = s1;
            part1[r] += (i1 == col) ? 0.0f : __expf(s1);
        }
    }

    // Reduce across the 16 lanes sharing each row; deterministic partial store.
    #pragma unroll
    for (int r = 0; r < 8; ++r) {
        float p0 = part0[r];
        float p1 = part1[r];
        #pragma unroll
        for (int m = 8; m >= 1; m >>= 1) {
            p0 += __shfl_xor(p0, m, 16);
            p1 += __shfl_xor(p1, m, 16);
        }
        if (nl == 0) {
            partials[wv * BN + rb + r + 8 * half]      = p0;
            partials[wv * BN + rb + 16 + r + 8 * half] = p1;
        }
    }
}

// ---------------- kernel 3: final deterministic reduction -------------------
__global__ __launch_bounds__(256)
void k_finalize(const float* __restrict__ partials,
                const float* __restrict__ rowpos,
                float* __restrict__ out) {
    const int t = threadIdx.x;
    float s = 0.0f;
    for (int i = t; i < BN; i += 256) {
        float rs = 0.0f;
        #pragma unroll
        for (int w = 0; w < 8; ++w) rs += partials[w * BN + i];
        s += logf(rs) - rowpos[i];
    }
    #pragma unroll
    for (int m = 16; m >= 1; m >>= 1) s += __shfl_xor(s, m, 32);
    __shared__ float red[8];
    if ((t & 31) == 0) red[t >> 5] = s;
    __syncthreads();
    if (t == 0) {
        float tot = 0.0f;
        #pragma unroll
        for (int w = 0; w < 8; ++w) tot += red[w];
        out[0] = tot / (float)BN;
    }
}

extern "C" void kernel_launch(void* const* d_in, const int* in_sizes, int n_in,
                              void* d_out, int out_size, void* d_ws, size_t ws_size,
                              hipStream_t stream) {
    const float* X = (const float*)d_in[0];

    _Float16* Xhi = (_Float16*)d_ws;                          // 8 MB
    _Float16* Xlo = Xhi + (size_t)BN * DK;                    // 8 MB
    float* partials = (float*)((char*)d_ws +
                      (size_t)2 * BN * DK * sizeof(_Float16)); // 256 KB
    float* rowpos = partials + 8 * BN;                        // 32 KB

    k_norm_split<<<BN, 128, 0, stream>>>(X, Xhi, Xlo);
    k_sim_lse<<<BN / 32, 256, 0, stream>>>(Xhi, Xlo, partials, rowpos);
    k_finalize<<<1, 256, 0, stream>>>(partials, rowpos, (float*)d_out);
}
